// TreeCRF_34857954574885
// MI455X (gfx1250) — matrix operand order repaired and checked by reading, not compiled, not run on hardware
//
#include <hip/hip_runtime.h>
#include <math.h>

typedef __attribute__((ext_vector_type(16))) _Float16 v16h;
typedef __attribute__((ext_vector_type(8)))  _Float16 v8h;
typedef __attribute__((ext_vector_type(8)))  float    v8f;

#define WAVES_PER_BLOCK 8
#define BLOCK (WAVES_PER_BLOCK * 32)
#define PITCH 24   // halves per staged row: 48B pitch, keeps 16B alignment, spreads banks

// max over each 16-lane half of the wave32
__device__ __forceinline__ float halfmax16(float v) {
#pragma unroll
  for (int off = 1; off < 16; off <<= 1)
    v = fmaxf(v, __shfl_xor(v, off, 32));
  return v;
}

// Build A fragment: A[0][0:16] = wl, A[1][16:32] = wr, rest 0.
// 16-bit A layout: lane L: M=L&15, hb=L>>4; h[e<8] -> col 8*hb+e ; h[e>=8] -> col 16+8*hb+(e-8)
__device__ __forceinline__ v16h build_A(const _Float16* ws, int M, int hb) {
  v16h A;
#pragma unroll
  for (int e = 0; e < 16; ++e) A[e] = (_Float16)0.0f;
  if (M == 0) {
    v8h wlo = *(const v8h*)&ws[8 * hb];        // wl[8hb .. 8hb+7]  (cols < 16)
#pragma unroll
    for (int e = 0; e < 8; ++e) A[e] = wlo[e];
  } else if (M == 1) {
    v8h whi = *(const v8h*)&ws[16 + 8 * hb];   // wr[8hb .. 8hb+7]  (cols >= 16)
#pragma unroll
    for (int e = 0; e < 8; ++e) A[8 + e] = whi[e];
  }
  return A;
}

// Build B fragment from staged rows: B[8*hb+e][n] = stage[n*PITCH + 8*hb+e], B[16+8*hb+e][n] = stage[(16+n)*PITCH + 8*hb+e]
__device__ __forceinline__ v16h build_B(const _Float16* stage, int n, int hb) {
  v8h b0 = *(const v8h*)&stage[n * PITCH + 8 * hb];
  v8h b1 = *(const v8h*)&stage[(16 + n) * PITCH + 8 * hb];
  v16h B;
#pragma unroll
  for (int e = 0; e < 8; ++e) { B[e] = b0[e]; B[8 + e] = b1[e]; }
  return B;
}

__global__ void bp_init(const float* __restrict__ unary, float* __restrict__ subtotal, int total) {
  int i = blockIdx.x * blockDim.x + threadIdx.x;
  if (i < total) subtotal[i] = unary[i];
}

__global__ void bp_root(const float* __restrict__ subtotal, float* __restrict__ out) {
  int i = threadIdx.x;
  if (i < 16) out[i] = subtotal[i];   // m_down(root) = 0
}

// -------- Upward pass: one wave handles one sibling pair (vl=2p+1, vr=2p+2) --------
__global__ __launch_bounds__(BLOCK) void bp_up(
    const float* __restrict__ edge_pot, float* __restrict__ subtotal,
    float* __restrict__ m_up, int pbase, int npairs)
{
  __shared__ __align__(16) _Float16 Es[WAVES_PER_BLOCK][2 * 16 * PITCH];
  __shared__ __align__(16) _Float16 Ws[WAVES_PER_BLOCK][32];

  const int lane = threadIdx.x & 31;
  const int wv   = threadIdx.x >> 5;
  const int q    = blockIdx.x * WAVES_PER_BLOCK + wv;
  if (q >= npairs) return;

  const int p  = pbase + q;
  const int vl = 2 * p + 1;
  const int n  = lane & 15;
  const int hb = lane >> 4;

  // weights: lanes 0-15 left child, 16-31 right child (contiguous 128B load)
  float s  = subtotal[(size_t)vl * 16 + lane];
  float mx = halfmax16(s);
  Ws[wv][lane] = (_Float16)__expf(s - mx);

  // stage exp(edge_pot): lane covers row n of child hb; pair = contiguous 2KB
  {
    const float4* src = (const float4*)(edge_pot + (size_t)(vl - 1) * 256);
    _Float16* rp = &Es[wv][(hb * 16 + n) * PITCH];
#pragma unroll
    for (int c = 0; c < 4; ++c) {
      float4 v = src[lane * 4 + c];
      rp[c * 4 + 0] = (_Float16)__expf(v.x);
      rp[c * 4 + 1] = (_Float16)__expf(v.y);
      rp[c * 4 + 2] = (_Float16)__expf(v.z);
      rp[c * 4 + 3] = (_Float16)__expf(v.w);
    }
  }
  __asm__ volatile("" ::: "memory");  // LDS pipe is in-order within a wave; just pin program order

  v16h A = build_A(&Ws[wv][0], n, hb);
  v16h B = build_B(&Es[wv][0], n, hb);
  v8f  c = {};
  v8f  D = __builtin_amdgcn_wmma_f32_16x16x32_f16(false, A, false, B, (short)0, c, false, false);

  float mxo = __shfl_xor(mx, 16, 32);  // other half's max
  if (hb == 0) {                        // lanes 0-15: D[0][n]=S_l[n], D[1][n]=S_r[n]
    float ml = mx  + __logf(D[0]);
    float mr = mxo + __logf(D[1]);
    m_up[(size_t)vl * 16 + n]       = ml;
    m_up[(size_t)(vl + 1) * 16 + n] = mr;
    subtotal[(size_t)p * 16 + n]   += ml + mr;   // unique writer per parent
  }
}

// -------- Downward pass: out[p] already holds subtotal[p]+m_down[p] --------
__global__ __launch_bounds__(BLOCK) void bp_down(
    const float* __restrict__ edge_pot, const float* __restrict__ subtotal,
    const float* __restrict__ m_up, float* __restrict__ out,
    int pbase, int npairs)
{
  __shared__ __align__(16) _Float16 Es[WAVES_PER_BLOCK][2 * 16 * PITCH];
  __shared__ __align__(16) _Float16 Ws[WAVES_PER_BLOCK][32];

  const int lane = threadIdx.x & 31;
  const int wv   = threadIdx.x >> 5;
  const int q    = blockIdx.x * WAVES_PER_BLOCK + wv;
  if (q >= npairs) return;

  const int p  = pbase + q;
  const int vl = 2 * p + 1;
  const int n  = lane & 15;
  const int hb = lane >> 4;

  // v2f = (subtotal[p] + m_down[p]) - m_up[child] = out[p] - m_up[child]
  float op  = out[(size_t)p * 16 + n];
  float mu  = m_up[(size_t)vl * 16 + lane];
  float v2f = op - mu;
  float mx  = halfmax16(v2f);
  Ws[wv][lane] = (_Float16)__expf(v2f - mx);

  // stage exp(ep^T): Es[hb][n][k] = exp(ep_hb[k][n]); per-k loads coalesce across lanes
  {
    const float* src = edge_pot + (size_t)(vl - 1) * 256 + (size_t)hb * 256 + n;
    _Float16* rp = &Es[wv][(hb * 16 + n) * PITCH];
#pragma unroll
    for (int k = 0; k < 16; ++k) rp[k] = (_Float16)__expf(src[k * 16]);
  }
  __asm__ volatile("" ::: "memory");

  v16h A = build_A(&Ws[wv][0], n, hb);
  v16h B = build_B(&Es[wv][0], n, hb);
  v8f  c = {};
  v8f  D = __builtin_amdgcn_wmma_f32_16x16x32_f16(false, A, false, B, (short)0, c, false, false);

  float mxo = __shfl_xor(mx, 16, 32);
  if (hb == 0) {
    float dl = mx  + __logf(D[0]);   // m_down[vl][n]
    float dr = mxo + __logf(D[1]);   // m_down[vr][n]
    out[(size_t)vl * 16 + n]       = subtotal[(size_t)vl * 16 + n]       + dl;
    out[(size_t)(vl + 1) * 16 + n] = subtotal[(size_t)(vl + 1) * 16 + n] + dr;
  }
}

extern "C" void kernel_launch(void* const* d_in, const int* in_sizes, int n_in,
                              void* d_out, int out_size, void* d_ws, size_t ws_size,
                              hipStream_t stream) {
  (void)in_sizes; (void)n_in; (void)out_size; (void)ws_size;
  const float* unary = (const float*)d_in[0];
  const float* edge  = (const float*)d_in[1];
  const int   DEPTH  = 18;                       // reference compile-time constant
  const size_t N     = ((size_t)1 << (DEPTH + 1)) - 1;

  float* subtotal = (float*)d_ws;                // N*16 floats
  float* m_up     = subtotal + N * 16;           // N*16 floats
  float* out      = (float*)d_out;

  const int total = (int)(N * 16);
  bp_init<<<(total + 255) / 256, 256, 0, stream>>>(unary, subtotal, total);

  for (int d = DEPTH; d >= 1; --d) {
    int npairs = 1 << (d - 1);
    int pbase  = (1 << (d - 1)) - 1;
    int grid   = (npairs + WAVES_PER_BLOCK - 1) / WAVES_PER_BLOCK;
    bp_up<<<grid, BLOCK, 0, stream>>>(edge, subtotal, m_up, pbase, npairs);
  }

  bp_root<<<1, 32, 0, stream>>>(subtotal, out);

  for (int d = 1; d <= DEPTH; ++d) {
    int npairs = 1 << (d - 1);
    int pbase  = (1 << (d - 1)) - 1;
    int grid   = (npairs + WAVES_PER_BLOCK - 1) / WAVES_PER_BLOCK;
    bp_down<<<grid, BLOCK, 0, stream>>>(edge, subtotal, m_up, out, pbase, npairs);
  }
}
